// LocalAttentionBlock_74775380623593
// MI455X (gfx1250) — compile-verified, matrix operand order checked by reading, not run on hardware
//
#include <hip/hip_runtime.h>

#define S_LEN 4096
#define DM    1024
#define NHEAD 16
#define DKH   64

typedef unsigned short u16;
typedef unsigned int   u32;
typedef __attribute__((ext_vector_type(16))) __bf16 v16bf;
typedef __attribute__((ext_vector_type(8)))  __bf16 v8bf;
typedef __attribute__((ext_vector_type(8)))  float  v8f;
typedef __attribute__((ext_vector_type(4)))  u32    v4u;
typedef __attribute__((ext_vector_type(8)))  int    v8i;
typedef __attribute__((ext_vector_type(4)))  int    v4i;

#if __has_builtin(__builtin_amdgcn_tensor_load_to_lds) && \
    __has_builtin(__builtin_amdgcn_s_wait_tensorcnt)
#define HAVE_TDM 1
#else
#define HAVE_TDM 0
#endif

__device__ __forceinline__ u16 f2bf_u(float x) {
  u32 u = __builtin_bit_cast(u32, x);
  u32 r = (u + 0x7FFFu + ((u >> 16) & 1u)) >> 16;   // RNE truncate to bf16
  return (u16)r;
}
__device__ __forceinline__ __bf16 f2bf(float x) {
  u16 r = f2bf_u(x);
  return __builtin_bit_cast(__bf16, r);
}
__device__ __forceinline__ v16bf cat16(v8bf lo, v8bf hi) {
  return __builtin_shufflevector(lo, hi, 0,1,2,3,4,5,6,7,8,9,10,11,12,13,14,15);
}
__device__ __forceinline__ v8bf ldg8(const u16* p) { return *(const v8bf*)p; }

__device__ __forceinline__ v8f wmma_bf16(v16bf a, v16bf b, v8f c) {
  // D = A(16x32) * B(32x16) + C, f32 accumulate
  return __builtin_amdgcn_wmma_f32_16x16x32_bf16(false, a, false, b, (short)0, c, false, false);
}

// ---------------------------------------------------------------------------
// fp32 -> bf16 conversion (bandwidth-bound, ~7us for all tensors @23.3TB/s)
// ---------------------------------------------------------------------------
__global__ void __launch_bounds__(256)
cvt_bf16_kernel(const float* __restrict__ src, u16* __restrict__ dst, int n4) {
  int i = blockIdx.x * 256 + threadIdx.x;
  if (i < n4) {
    float4 f = ((const float4*)src)[i];
    u32 lo = f2bf_u(f.x) | ((u32)f2bf_u(f.y) << 16);
    u32 hi = f2bf_u(f.z) | ((u32)f2bf_u(f.w) << 16);
    ((uint2*)dst)[i] = make_uint2(lo, hi);
  }
}

// ---------------------------------------------------------------------------
// TDM: DMA a [rows=128 x cols=64] bf16 tile (row stride 1024 elems) into LDS
// with 16B padding after each 128B row -> LDS pitch 72 halfwords (bank-safe).
// D# per CDNA5 ISA ch.8: group0 {count=1, lds_addr, global_addr, type=2},
// group1 {data_size=2B, pad_en, pad_interval=32dw, pad_amount=4dw, dims}.
// ---------------------------------------------------------------------------
#if HAVE_TDM
__device__ __forceinline__ void tdm_load_tile(const u16* gsrc, u32 lds_off,
                                              u32 tensor_rows) {
  unsigned long long ga = (unsigned long long)(uintptr_t)gsrc;
  v4u g0;
  g0[0] = 1u;                                   // count=1 (valid descriptor)
  g0[1] = lds_off;                              // lds_addr (bytes)
  g0[2] = (u32)ga;                              // global_addr[31:0]
  g0[3] = (u32)((ga >> 32) & 0x1FFFFFFu) | (2u << 30);  // addr[56:32] | type=2
  v8i g1;
  g1[0] = (int)((1u << 16)      // data_size = 2 bytes
              | (1u << 20)      // pad_enable
              | (4u << 22)      // pad_interval: 32 DWORDs (=128B = one row)
              | (3u << 25));    // pad_amount: 4 DWORDs (=16B = 8 halfwords)
  g1[1] = (int)(1024u << 16);                   // tensor_dim0 = 1024 (lo16)
  g1[2] = (int)((tensor_rows & 0xFFFFu) << 16); // tensor_dim1 lo16
  g1[3] = (int)(64u << 16);                     // tensor_dim1 hi16=0 | tile_dim0=64
  g1[4] = (int)128;                             // tile_dim1 = 128, tile_dim2 = 0
  g1[5] = (int)1024;                            // tensor_dim0_stride = 1024
  g1[6] = 0;
  g1[7] = 0;
  v4i gz; gz[0] = 0; gz[1] = 0; gz[2] = 0; gz[3] = 0;   // 2-D tensor: groups 2/3 unused
#if __clang_major__ >= 23
  v8i z8; z8[0]=0; z8[1]=0; z8[2]=0; z8[3]=0; z8[4]=0; z8[5]=0; z8[6]=0; z8[7]=0;
  __builtin_amdgcn_tensor_load_to_lds(g0, g1, gz, gz, z8, 0);
#else
  __builtin_amdgcn_tensor_load_to_lds(g0, g1, gz, gz, 0);
#endif
}
#endif

// ---------------------------------------------------------------------------
// GEMM: out = A[M=8192,K=1024](bf16) @ W[N=1024,K=1024](bf16)^T
// MODE 0: store bf16 into [b*H+h][s][64]       (Q, K heads)
// MODE 1: store bf16 into [b*H+h][64][s]       (V transposed)
// MODE 2: store f32  into [m][1024]            (final output)
// ---------------------------------------------------------------------------
template<int MODE>
__global__ void __launch_bounds__(256)
gemm_kernel(const u16* __restrict__ Ab, const u16* __restrict__ Bb,
            void* __restrict__ Cp)
{
  __shared__ __attribute__((aligned(16))) u16 As[2][128 * 72];
  __shared__ __attribute__((aligned(16))) u16 Bs[2][128 * 72];

  const int tid  = threadIdx.x;
  const int wave = tid >> 5;
  const int lane = tid & 31;
  const int wm   = wave >> 2;        // 0..1  (64-row slab)
  const int wn   = wave & 3;         // 0..3  (32-col slab)
  const int lx   = lane & 15;
  const int hb   = (lane >> 4) * 8;  // K sub-offset per half-wave
  const long mblk = (long)blockIdx.x * 128;
  const int  nblk = blockIdx.y * 128;

  v8f acc[4][2];
#pragma unroll
  for (int i = 0; i < 4; ++i)
#pragma unroll
    for (int j = 0; j < 2; ++j)
      acc[i][j] = (v8f){0,0,0,0,0,0,0,0};

#if HAVE_TDM
  // ---- double-buffered TDM pipeline: DMA step s+1 overlaps WMMA of step s --
  if (wave == 0) {
    tdm_load_tile(Ab + mblk * DM, (u32)(uintptr_t)&As[0][0], 8192u);
    tdm_load_tile(Bb + (long)nblk * DM, (u32)(uintptr_t)&Bs[0][0], 1024u);
    __builtin_amdgcn_s_wait_tensorcnt(0);
  }
  __syncthreads();
  for (int step = 0; step < DM / 64; ++step) {
    const int buf = step & 1;
    if (wave == 0 && step + 1 < DM / 64) {
      const int k1 = (step + 1) * 64;
      tdm_load_tile(Ab + mblk * DM + k1, (u32)(uintptr_t)&As[buf ^ 1][0], 8192u);
      tdm_load_tile(Bb + (long)nblk * DM + k1, (u32)(uintptr_t)&Bs[buf ^ 1][0], 1024u);
    }
#else
  for (int step = 0; step < DM / 64; ++step) {
    const int buf = 0;
    const int k0 = step * 64;
    __syncthreads();
#pragma unroll
    for (int i = 0; i < 4; ++i) {
      int c = tid + i * 256;                   // 0..1023
      int row = c >> 3, col = (c & 7) * 8;
      *(uint4*)(&As[0][row * 72 + col]) =
          *(const uint4*)(Ab + (mblk + row) * (long)DM + k0 + col);
      *(uint4*)(&Bs[0][row * 72 + col]) =
          *(const uint4*)(Bb + (long)(nblk + row) * DM + k0 + col);
    }
    __syncthreads();
#endif

#pragma unroll
    for (int kc = 0; kc < 64; kc += 32) {
      v16bf a[4], b[2];
#pragma unroll
      for (int mf = 0; mf < 4; ++mf) {
        const u16* p = &As[buf][(wm * 64 + mf * 16 + lx) * 72 + kc + hb];
        a[mf] = cat16(*(const v8bf*)p, *(const v8bf*)(p + 16));
      }
#pragma unroll
      for (int nf = 0; nf < 2; ++nf) {
        const u16* p = &Bs[buf][(wn * 32 + nf * 16 + lx) * 72 + kc + hb];
        b[nf] = cat16(*(const v8bf*)p, *(const v8bf*)(p + 16));
      }
#pragma unroll
      for (int mf = 0; mf < 4; ++mf)
#pragma unroll
        for (int nf = 0; nf < 2; ++nf)
          acc[mf][nf] = wmma_bf16(a[mf], b[nf], acc[mf][nf]);
    }

#if HAVE_TDM
    if (wave == 0 && step + 1 < DM / 64)
      __builtin_amdgcn_s_wait_tensorcnt(0);    // next-step tiles fully in LDS
    __syncthreads();                           // publish to all waves
#endif
  }

  // ---- store ----
#pragma unroll
  for (int mf = 0; mf < 4; ++mf) {
#pragma unroll
    for (int nf = 0; nf < 2; ++nf) {
      v8f c = acc[mf][nf];
      int  n  = nblk + wn * 32 + nf * 16 + lx;
      long m0 = mblk + wm * 64 + mf * 16 + hb;   // + v
      if (MODE == 2) {
        float* O = (float*)Cp;
#pragma unroll
        for (int v = 0; v < 8; ++v)
          O[(m0 + v) * (long)DM + n] = c[v];
      } else {
        u16* O = (u16*)Cp;
        int  hh = n >> 6, dk = n & 63;
        long bb = m0 >> 12;                 // batch  (S=4096)
        long ss = m0 & 4095;                // seq pos
        if (MODE == 0) {
          long base = (((bb * NHEAD + hh) * S_LEN) + ss) * DKH + dk;
#pragma unroll
          for (int v = 0; v < 8; ++v)
            O[base + (long)v * DKH] = f2bf_u(c[v]);
        } else { // MODE 1: transposed V
          long base = ((bb * NHEAD + hh) * DKH + dk) * (long)S_LEN + ss;
          u32 w0 = f2bf_u(c[0]) | ((u32)f2bf_u(c[1]) << 16);
          u32 w1 = f2bf_u(c[2]) | ((u32)f2bf_u(c[3]) << 16);
          u32 w2 = f2bf_u(c[4]) | ((u32)f2bf_u(c[5]) << 16);
          u32 w3 = f2bf_u(c[6]) | ((u32)f2bf_u(c[7]) << 16);
          *(uint4*)(O + base) = make_uint4(w0, w1, w2, w3);
        }
      }
    }
  }
}

// ---------------------------------------------------------------------------
// Local attention: one wave = 16 queries of one (b,h). Computes S^T = K*Q^T so
// exp(S^T) IS the B-operand layout for the P*V WMMA (no LDS, no transposes).
// ---------------------------------------------------------------------------
__global__ void __launch_bounds__(256)
attn_kernel(const u16* __restrict__ Qh, const u16* __restrict__ Kh,
            const u16* __restrict__ Vt, u16* __restrict__ AO)
{
  const int bh   = blockIdx.y;
  const int b    = bh >> 4;
  const int h    = bh & 15;
  const int wave = threadIdx.x >> 5;
  const int lane = threadIdx.x & 31;
  const int lx   = lane & 15;
  const int hb   = (lane >> 4) * 8;
  const int qbase = blockIdx.x * 128 + wave * 16;
  const int q     = qbase + lx;

  // Q as B-operand: column q, d striped
  const u16* Qrow = Qh + ((long)(bh * S_LEN + q)) * DKH;
  v16bf bQ[2];
#pragma unroll
  for (int kd = 0; kd < 2; ++kd)
    bQ[kd] = cat16(ldg8(Qrow + kd * 32 + hb), ldg8(Qrow + kd * 32 + 16 + hb));

  float m = -1e9f, l = 0.f;
  v8f acc[4];
#pragma unroll
  for (int f = 0; f < 4; ++f) acc[f] = (v8f){0,0,0,0,0,0,0,0};

  const float CS = 0.125f * 1.44269504f;   // 1/sqrt(64) * log2(e)

  // valid keys for this lane's q form one contiguous range [lo, lo+span]
  const int lo   = q - 128;
  const int hi   = (q + 128 < S_LEN - 1) ? (q + 128) : (S_LEN - 1);
  const u32 span = (u32)(hi - lo);

  int kstart = qbase - 128; if (kstart < 0) kstart = 0;
  kstart &= ~31;
  int kend = qbase + 144; if (kend > S_LEN) kend = S_LEN;

  for (int kc = kstart; kc < kend; kc += 32) {
    // ---- S^T = K_chunk @ Q^T : two 16x16 C-frags (key halves) ----
    v8f s0 = (v8f){0,0,0,0,0,0,0,0};
    v8f s1 = (v8f){0,0,0,0,0,0,0,0};
    {
      int key0 = kc + lx;        if (key0 > S_LEN - 1) key0 = S_LEN - 1;
      int key1 = kc + 16 + lx;   if (key1 > S_LEN - 1) key1 = S_LEN - 1;
      const u16* K0 = Kh + ((long)(bh * S_LEN + key0)) * DKH;
      const u16* K1 = Kh + ((long)(bh * S_LEN + key1)) * DKH;
      v16bf a00 = cat16(ldg8(K0 + hb),      ldg8(K0 + 16 + hb));
      v16bf a01 = cat16(ldg8(K0 + 32 + hb), ldg8(K0 + 48 + hb));
      v16bf a10 = cat16(ldg8(K1 + hb),      ldg8(K1 + 16 + hb));
      v16bf a11 = cat16(ldg8(K1 + 32 + hb), ldg8(K1 + 48 + hb));
      s0 = wmma_bf16(a00, bQ[0], s0);
      s0 = wmma_bf16(a01, bQ[1], s0);
      s1 = wmma_bf16(a10, bQ[0], s1);
      s1 = wmma_bf16(a11, bQ[1], s1);
    }
    // ---- mask (single unsigned range-compare) + scale (base-2 domain) ----
    v8f y0, y1;
    float cm = -1e9f;
    const int rel = kc + hb - lo;          // chunk-relative key minus lo
#pragma unroll
    for (int v = 0; v < 8; ++v) {
      bool ok0 = (u32)(rel + v)      <= span;
      bool ok1 = (u32)(rel + v + 16) <= span;
      y0[v] = ok0 ? s0[v] * CS : -1e9f;
      y1[v] = ok1 ? s1[v] * CS : -1e9f;
      cm = fmaxf(cm, fmaxf(y0[v], y1[v]));
    }
    cm = fmaxf(cm, __shfl_xor(cm, 16, 32));    // combine key halves
    float mn   = fmaxf(m, cm);
    float corr = __builtin_amdgcn_exp2f(m - mn);
    float rs   = 0.f;
    v16bf Pt;
#pragma unroll
    for (int v = 0; v < 8; ++v) {
      float e0 = __builtin_amdgcn_exp2f(y0[v] - mn);
      float e1 = __builtin_amdgcn_exp2f(y1[v] - mn);
      rs += e0 + e1;
      Pt[v]     = f2bf(e0);   // keys hb+v       -> B-frag elems 0..7
      Pt[v + 8] = f2bf(e1);   // keys hb+16+v    -> B-frag elems 8..15
    }
    rs += __shfl_xor(rs, 16, 32);
    l = l * corr + rs;
    m = mn;
    // ---- O^T += V^T_chunk @ P^T ----
#pragma unroll
    for (int f = 0; f < 4; ++f) {
#pragma unroll
      for (int v = 0; v < 8; ++v) acc[f][v] *= corr;
      const u16* Vrow = Vt + ((long)(bh * DKH + f * 16 + lx)) * S_LEN + kc + hb;
      v16bf aV = cat16(ldg8(Vrow), ldg8(Vrow + 16));
      acc[f] = wmma_bf16(aV, Pt, acc[f]);
    }
  }

  const float rinv = 1.0f / l;
  u16* Ob = AO + ((long)(b * S_LEN + q)) * DM + h * DKH;
#pragma unroll
  for (int f = 0; f < 4; ++f) {
    v8f c = acc[f];
    u32 w0 = f2bf_u(c[0] * rinv) | ((u32)f2bf_u(c[1] * rinv) << 16);
    u32 w1 = f2bf_u(c[2] * rinv) | ((u32)f2bf_u(c[3] * rinv) << 16);
    u32 w2 = f2bf_u(c[4] * rinv) | ((u32)f2bf_u(c[5] * rinv) << 16);
    u32 w3 = f2bf_u(c[6] * rinv) | ((u32)f2bf_u(c[7] * rinv) << 16);
    *(uint4*)(Ob + f * 16 + hb) = make_uint4(w0, w1, w2, w3);
  }
}

// ---------------------------------------------------------------------------
extern "C" void kernel_launch(void* const* d_in, const int* in_sizes, int n_in,
                              void* d_out, int out_size, void* d_ws, size_t ws_size,
                              hipStream_t stream) {
  const float* q  = (const float*)d_in[0];
  const float* k  = (const float*)d_in[1];
  const float* v  = (const float*)d_in[2];
  const float* wq = (const float*)d_in[3];
  const float* wk = (const float*)d_in[4];
  const float* wv = (const float*)d_in[5];
  const float* wo = (const float*)d_in[6];

  const size_t BIG = (size_t)2 * NHEAD * S_LEN * DKH;   // 8,388,608 elems
  const size_t WSZ = (size_t)DM * DM;                   // 1,048,576 elems
  u16* Qh  = (u16*)d_ws;
  u16* Kh  = Qh  + BIG;
  u16* Vt  = Kh  + BIG;
  u16* AO  = Vt  + BIG;
  u16* qb  = AO  + BIG;
  u16* kb  = qb  + BIG;
  u16* vb  = kb  + BIG;
  u16* wqb = vb  + BIG;
  u16* wkb = wqb + WSZ;
  u16* wvb = wkb + WSZ;
  u16* wob = wvb + WSZ;

  dim3 blk(256);
  const int big4 = (int)(BIG / 4), w4 = (int)(WSZ / 4);
  cvt_bf16_kernel<<<big4 / 256, blk, 0, stream>>>(q,  qb,  big4);
  cvt_bf16_kernel<<<big4 / 256, blk, 0, stream>>>(k,  kb,  big4);
  cvt_bf16_kernel<<<big4 / 256, blk, 0, stream>>>(v,  vb,  big4);
  cvt_bf16_kernel<<<w4 / 256,   blk, 0, stream>>>(wq, wqb, w4);
  cvt_bf16_kernel<<<w4 / 256,   blk, 0, stream>>>(wk, wkb, w4);
  cvt_bf16_kernel<<<w4 / 256,   blk, 0, stream>>>(wv, wvb, w4);
  cvt_bf16_kernel<<<w4 / 256,   blk, 0, stream>>>(wo, wob, w4);

  dim3 gg(64, 8);                 // 8192/128 x 1024/128
  dim3 ga(S_LEN / 128, 2 * NHEAD);
  gemm_kernel<0><<<gg, blk, 0, stream>>>(qb, wqb, Qh);
  gemm_kernel<0><<<gg, blk, 0, stream>>>(kb, wkb, Kh);
  gemm_kernel<1><<<gg, blk, 0, stream>>>(vb, wvb, Vt);
  attn_kernel<<<ga, blk, 0, stream>>>(Qh, Kh, Vt, AO);
  gemm_kernel<2><<<gg, blk, 0, stream>>>(AO, wob, d_out);
}
